// SSD_R34_42288247996610
// MI455X (gfx1250) — compile-verified, matrix operand order checked by reading, not run on hardware
//
#include <hip/hip_runtime.h>
#include <stdint.h>

#define BATCH 32
#define NCLS 81
#define NKEEP 80          // classes 1..80 (class 0 dropped)
#define MAXD 200
#define IOU_T 0.5f
#define CONF_T 0.05f
#define TILES 13          // ceil(208/16) covering 200, padded
#define PAD 208

typedef __attribute__((ext_vector_type(2)))  float        v2f;
typedef __attribute__((ext_vector_type(8)))  float        v8f;
typedef __attribute__((ext_vector_type(16))) _Float16     v16h;
typedef __attribute__((ext_vector_type(4)))  unsigned int u32x4;
typedef __attribute__((ext_vector_type(4)))  int          i32x4;
typedef __attribute__((ext_vector_type(8)))  int          i32x8;

#if __has_builtin(__builtin_amdgcn_tensor_load_to_lds)
#define HAVE_TDM 1
#else
#define HAVE_TDM 0
#endif

// ---------------------------------------------------------------------------
// K1: decode boxes to LTRB + softmax over 81 classes, streaming & coalesced.
// ---------------------------------------------------------------------------
__global__ void ssd_decode_softmax(const float* __restrict__ bbin,
                                   const float* __restrict__ scin,
                                   const float* __restrict__ sxy_p,
                                   const float* __restrict__ swh_p,
                                   const float* __restrict__ db,
                                   float* __restrict__ ws_boxes,
                                   float* __restrict__ ws_cls,
                                   int N) {
    int n = blockIdx.x * blockDim.x + threadIdx.x;
    int b = blockIdx.y;
    if (n >= N) return;
    float sxy = sxy_p[0], swh = swh_p[0];

    const float* bb = bbin + (size_t)b * 4 * N;
    float bx = bb[(size_t)0 * N + n], by = bb[(size_t)1 * N + n];
    float bw = bb[(size_t)2 * N + n], bh = bb[(size_t)3 * N + n];
    float dx = db[n * 4 + 0], dy = db[n * 4 + 1];
    float dw = db[n * 4 + 2], dh = db[n * 4 + 3];

    float x = bx * sxy * dw + dx;
    float y = by * sxy * dh + dy;
    float w = __expf(bw * swh) * dw;
    float h = __expf(bh * swh) * dh;

    float* ob = ws_boxes + ((size_t)b * N + n) * 4;
    ob[0] = x - 0.5f * w;
    ob[1] = y - 0.5f * h;
    ob[2] = x + 0.5f * w;
    ob[3] = y + 0.5f * h;

    const float* sc = scin + (size_t)b * NCLS * N + n;
    float m = -1e30f;
    for (int c = 0; c < NCLS; ++c) {
        if (c + 4 < NCLS) __builtin_prefetch(sc + (size_t)(c + 4) * N, 0, 1);
        m = fmaxf(m, sc[(size_t)c * N]);
    }
    float sum = 0.f;
    for (int c = 0; c < NCLS; ++c) sum += __expf(sc[(size_t)c * N] - m);
    float inv = 1.f / sum;
    for (int c = 1; c < NCLS; ++c)
        ws_cls[((size_t)b * NKEEP + (c - 1)) * N + n] =
            __expf(sc[(size_t)c * N] - m) * inv;
}

// ---------------------------------------------------------------------------
// K2 (fused top-k + NMS): per (batch, class).
//  - Stage the whole 15130-float class row into LDS with one TDM descriptor
//    (tensor_load_to_lds + s_wait_tensorcnt), fallback: coalesced loads.
//  - 200x argmax entirely from LDS (wave32 shfl reduce + 8-slot merge).
//  - NMS: area_i+area_j per 16x16 tile via V_WMMA_F32_16X16X4_F32,
//    intersection via VALU, suppression bitmask via DS atomics,
//    sequential keep loop, then emit top-200 boxes/scores for the merge.
// ---------------------------------------------------------------------------
__global__ void ssd_topk_nms(const float* __restrict__ ws_boxes,
                             const float* __restrict__ ws_cls,
                             float* __restrict__ tb,
                             float* __restrict__ ts,
                             int N) {
    extern __shared__ float sscores[];       // N floats, dynamic LDS
    __shared__ float    sb[PAD * 4];
    __shared__ float    sarea[PAD];
    __shared__ unsigned sup[MAXD][7];
    __shared__ unsigned keepw[7];
    __shared__ int      selidx[MAXD];
    __shared__ float    selval[MAXD];
    __shared__ float    wv[8];
    __shared__ int      wi[8];

    int bc = blockIdx.x;                     // b*80 + c
    int b = bc / NKEEP;
    int tid = threadIdx.x;
    int lane = tid & 31;
    int wave = tid >> 5;

    const float* arrg = ws_cls + (size_t)bc * N;

    // ---- stage class scores to LDS ----
#if HAVE_TDM
    if (wave == 0) {                         // one TDM issue per block
        unsigned long long ga = (unsigned long long)(uintptr_t)arrg;
        unsigned lds_off = (unsigned)(uintptr_t)sscores; // flat LDS addr[31:0] = LDS byte offset
        unsigned un = (unsigned)N;
        u32x4 g0 = { 1u,                                  // count=1, user desc
                     lds_off,                             // lds_addr
                     (unsigned)(ga & 0xFFFFFFFFu),        // global_addr[31:0]
                     (unsigned)((ga >> 32) & 0x01FFFFFFu) | (2u << 30) }; // [56:32] | type=2
        i32x8 g1 = { (int)(2u << 16),                     // data_size=4B, mask=0
                     (int)((un & 0xFFFFu) << 16),         // tensor_dim0[15:0]
                     (int)((un >> 16) | (1u << 16)),      // tensor_dim0[31:16] | tensor_dim1[15:0]=1
                     (int)((un & 0xFFFFu) << 16),         // tensor_dim1 hi=0 | tile_dim0=N
                     1,                                   // tile_dim1=1, tile_dim2=0
                     (int)un,                             // tensor_dim0_stride[31:0]
                     (int)((un & 0xFFFFu) << 16),         // stride0 hi=0 | stride1[15:0]=N
                     0 };                                 // stride1 hi
        i32x4 g2 = { 0, 0, 0, 0 };
        i32x4 g3 = { 0, 0, 0, 0 };
#if defined(__clang_major__) && __clang_major__ >= 23
        i32x8 g4 = { 0, 0, 0, 0, 0, 0, 0, 0 };
        __builtin_amdgcn_tensor_load_to_lds(g0, g1, g2, g3, g4, 0);
#else
        __builtin_amdgcn_tensor_load_to_lds(g0, g1, g2, g3, 0);
#endif
        __builtin_amdgcn_s_wait_tensorcnt(0);
    }
#else
    for (int n = tid; n < N; n += 256) sscores[n] = arrg[n];
#endif
    __syncthreads();

    // ---- top-200 via repeated argmax from LDS ----
    for (int k = 0; k < MAXD; ++k) {
        float best = -3.f;
        int bi = 0;
        for (int n = tid; n < N; n += 256) {
            float v = sscores[n];
            if (v > best) { best = v; bi = n; }
        }
        for (int off = 16; off > 0; off >>= 1) {
            float ov = __shfl_down(best, off, 32);
            int   oi = __shfl_down(bi, off, 32);
            if (ov > best || (ov == best && oi < bi)) { best = ov; bi = oi; }
        }
        if (lane == 0) { wv[wave] = best; wi[wave] = bi; }
        __syncthreads();
        if (tid == 0) {
            float bv = wv[0]; int bix = wi[0];
            for (int w = 1; w < 8; ++w) {
                float ov = wv[w]; int oi = wi[w];
                if (ov > bv || (ov == bv && oi < bix)) { bv = ov; bix = oi; }
            }
            selidx[k] = bix;
            selval[k] = (bv > CONF_T) ? bv : -1.f;
            sscores[bix] = -2.f;              // consume
        }
        __syncthreads();
    }

    // ---- stage selected boxes (padded to 208) + areas ----
    for (int i = tid; i < PAD; i += 256) {
        float l = 0.f, t = 0.f, r = 0.f, bo = 0.f;
        if (i < MAXD) {
            const float* src = ws_boxes + ((size_t)b * N + selidx[i]) * 4;
            l = src[0]; t = src[1]; r = src[2]; bo = src[3];
        }
        sb[i * 4 + 0] = l; sb[i * 4 + 1] = t;
        sb[i * 4 + 2] = r; sb[i * 4 + 3] = bo;
        sarea[i] = (r - l) * (bo - t);
    }
    for (int i = tid; i < MAXD; i += 256)
        for (int w = 0; w < 7; ++w) sup[i][w] = 0u;
    __syncthreads();

    // ---- pairwise IoU: WMMA for area sums, VALU for intersection ----
    bool lo = lane < 16;
    int m = lane & 15;
    for (int t = wave; t < TILES * TILES; t += 8) {   // wave-uniform control flow
        int ti = t / TILES, tj = t % TILES;
        float ai = lo ? sarea[ti * 16 + m] : 0.f;
        float aj = lo ? sarea[tj * 16 + m] : 0.f;
        v8f c = {};
#if __has_builtin(__builtin_amdgcn_wmma_f32_16x16x4_f32)
        v2f a, bv2;
        a[0]   = ai;             a[1]   = lo ? 1.f : 0.f;  // A[m][0]=area_i, A[m][1]=1
        bv2[0] = lo ? 1.f : 0.f; bv2[1] = aj;              // B[0][n]=1, B[1][n]=area_j
        c = __builtin_amdgcn_wmma_f32_16x16x4_f32(false, a, false, bv2,
                                                  (short)0, c, false, false);
#else
        v16h ah = {}, bh = {};
        ah[0] = (_Float16)ai;                       ah[1] = lo ? (_Float16)1.f : (_Float16)0.f;
        bh[0] = lo ? (_Float16)1.f : (_Float16)0.f; bh[1] = (_Float16)aj;
        c = __builtin_amdgcn_wmma_f32_16x16x32_f16(false, ah, false, bh,
                                                   (short)0, c, false, false);
#endif
        int j = tj * 16 + m;
#pragma unroll
        for (int v = 0; v < 8; ++v) {
            int i = ti * 16 + v + (lo ? 0 : 8);      // C/D layout: M=v+8*hi, N=lane%16
            if (i < MAXD && j < MAXD && j > i) {
                float lt0 = fmaxf(sb[i * 4 + 0], sb[j * 4 + 0]);
                float lt1 = fmaxf(sb[i * 4 + 1], sb[j * 4 + 1]);
                float rb0 = fminf(sb[i * 4 + 2], sb[j * 4 + 2]);
                float rb1 = fminf(sb[i * 4 + 3], sb[j * 4 + 3]);
                float w0 = fmaxf(rb0 - lt0, 0.f);
                float h0 = fmaxf(rb1 - lt1, 0.f);
                float inter = w0 * h0;
                float denom = c[v] - inter;          // area_i + area_j - inter
                if (denom > 0.f && inter > IOU_T * denom)
                    atomicOr(&sup[i][j >> 5], 1u << (j & 31));
            }
        }
    }
    __syncthreads();

    // ---- sequential suppression ----
    if (tid == 0) {
        unsigned kw[7];
        for (int w = 0; w < 7; ++w) kw[w] = 0u;
        for (int i = 0; i < MAXD; ++i)
            if (selval[i] > CONF_T) kw[i >> 5] |= 1u << (i & 31);
        for (int i = 0; i < MAXD; ++i) {
            if ((kw[i >> 5] >> (i & 31)) & 1u) {
                for (int w = 0; w < 7; ++w) kw[w] &= ~sup[i][w];  // sup[i] only has j>i
            }
        }
        for (int w = 0; w < 7; ++w) keepw[w] = kw[w];
    }
    __syncthreads();

    // ---- emit top-200 candidates for the final merge ----
    for (int i = tid; i < MAXD; i += 256) {
        bool kept = (keepw[i >> 5] >> (i & 31)) & 1u;
        ts[(size_t)bc * MAXD + i] = kept ? selval[i] : -1.f;
        float* dst = tb + ((size_t)bc * MAXD + i) * 4;
        dst[0] = sb[i * 4 + 0];
        dst[1] = sb[i * 4 + 1];
        dst[2] = sb[i * 4 + 2];
        dst[3] = sb[i * 4 + 3];
    }
}

// ---------------------------------------------------------------------------
// K3: per-batch final top-200 over 80*200 candidates; emit boxes/labels/scores.
// ---------------------------------------------------------------------------
__global__ void ssd_final_topk(const float* __restrict__ tb,
                               float* __restrict__ ts,
                               float* __restrict__ outb,
                               int* __restrict__ outl,
                               float* __restrict__ outs) {
    __shared__ float sval[256];
    __shared__ int   sidx[256];
    __shared__ int   selidx[MAXD];
    __shared__ float selval[MAXD];

    const int M = NKEEP * MAXD;   // 16000
    int b = blockIdx.x;
    int tid = threadIdx.x;
    float* arr = ts + (size_t)b * M;

    for (int k = 0; k < MAXD; ++k) {
        float best = -4.f;
        int bi = 0;
        for (int n = tid; n < M; n += 256) {
            float v = arr[n];
            if (v > best) { best = v; bi = n; }
        }
        sval[tid] = best; sidx[tid] = bi;
        __syncthreads();
        for (int off = 128; off > 0; off >>= 1) {
            if (tid < off) {
                float ov = sval[tid + off]; int oi = sidx[tid + off];
                if (ov > sval[tid] || (ov == sval[tid] && oi < sidx[tid])) {
                    sval[tid] = ov; sidx[tid] = oi;
                }
            }
            __syncthreads();
        }
        if (tid == 0) {
            selidx[k] = sidx[0];
            selval[k] = sval[0];
            arr[sidx[0]] = -3.f;   // consume
        }
        __syncthreads();
    }
    for (int k = tid; k < MAXD; k += 256) {
        int idx = selidx[k];
        float v = selval[k];
        bool valid = v > CONF_T;
        const float* src = tb + ((size_t)b * M + idx) * 4;
        float* dst = outb + ((size_t)b * MAXD + k) * 4;
        dst[0] = valid ? src[0] : 0.f;
        dst[1] = valid ? src[1] : 0.f;
        dst[2] = valid ? src[2] : 0.f;
        dst[3] = valid ? src[3] : 0.f;
        outl[b * MAXD + k] = valid ? (idx / MAXD + 1) : 0;
        outs[b * MAXD + k] = valid ? v : 0.f;
    }
}

// ---------------------------------------------------------------------------
extern "C" void kernel_launch(void* const* d_in, const int* in_sizes, int n_in,
                              void* d_out, int out_size, void* d_ws, size_t ws_size,
                              hipStream_t stream) {
    const float* bbin  = (const float*)d_in[0];   // (B,4,N)
    const float* scin  = (const float*)d_in[1];   // (B,81,N)
    const float* sxy   = (const float*)d_in[2];   // (1,)
    const float* swh   = (const float*)d_in[3];   // (1,)
    const float* db    = (const float*)d_in[4];   // (1,N,4)
    const int N = in_sizes[4] / 4;

    float* ws       = (float*)d_ws;
    float* ws_boxes = ws;                                          // B*N*4
    float* ws_cls   = ws_boxes + (size_t)BATCH * N * 4;            // B*80*N
    float* ws_tb    = ws_cls   + (size_t)BATCH * NKEEP * N;        // B*80*200*4
    float* ws_ts    = ws_tb    + (size_t)BATCH * NKEEP * MAXD * 4; // B*80*200

    float* outb = (float*)d_out;                           // B*200*4 floats
    int*   outl = (int*)(outb + (size_t)BATCH * MAXD * 4); // B*200 int32
    float* outs = (float*)(outl + (size_t)BATCH * MAXD);   // B*200 floats

    dim3 g1((N + 255) / 256, BATCH);
    ssd_decode_softmax<<<g1, 256, 0, stream>>>(bbin, scin, sxy, swh, db,
                                               ws_boxes, ws_cls, N);
    size_t dyn = (size_t)N * sizeof(float);   // 60.5 KB class row in LDS
    ssd_topk_nms<<<BATCH * NKEEP, 256, dyn, stream>>>(ws_boxes, ws_cls,
                                                      ws_tb, ws_ts, N);
    ssd_final_topk<<<BATCH, 256, 0, stream>>>(ws_tb, ws_ts, outb, outl, outs);
}
